// HypergraphGNN_39298950759065
// MI455X (gfx1250) — matrix operand order repaired and checked by reading, not compiled, not run on hardware
//
#include <hip/hip_runtime.h>
#include <cstddef>

// ---------------------------------------------------------------------------
// HypergraphGNN on MI455X (gfx1250, wave32, WMMA bf16 -> f32 accumulate)
//
// Algebraic rewrite: hv[src] @ W == (hv @ W)[src]  -> transform nodes, not
// edges (8x FLOP cut). Per-edge bias folded into segment mean via degree.
// Working set (~130MB) fits in 192MB L2, so scatter runs at L2 bandwidth.
// This rev: hardware f32->bf16 conversion (compiler-lowered casts) instead of
// the 3-VALU/elt software RNE trick, to raise WMMA density in the inner loop.
// ---------------------------------------------------------------------------

constexpr int kNV = 100000;
constexpr int kNC = 50000;
constexpr int kNE = 800000;
constexpr int kH  = 128;
constexpr float kLnEps  = 1e-5f;
constexpr float kCntEps = 1e-6f;
constexpr int kFragElems = 4 * 8 * 32 * 16;   // [kc][nt][lane][e] bf16 elems

typedef __attribute__((ext_vector_type(16))) __bf16 v16bf;
typedef __attribute__((ext_vector_type(8)))  float  v8f;

union Frag {
  v16bf v;
  unsigned short s[16];
  unsigned int   u[8];
};

// Pre-pack a 128x128 f32 row-major weight into bf16 B-fragments.
// B layout (contiguous k order, same as load_afrag): lane holds column
// n = nt*16 + (lane&15); element e -> k = kc*32 + (lane>>4)*16 + e.
__global__ void prep_wfrag(const float* __restrict__ w,
                           unsigned short* __restrict__ dst) {
  int t = blockIdx.x * blockDim.x + threadIdx.x;
  if (t >= kFragElems) return;
  int e    = t & 15;
  int lane = (t >> 4) & 31;
  int nt   = (t >> 9) & 7;
  int kc   = t >> 12;
  int k = kc * 32 + ((lane >> 4) << 4) + e;
  int n = nt * 16 + (lane & 15);
  __bf16 h = (__bf16)w[k * kH + n];            // hw cvt, RNE
  dst[t] = __builtin_bit_cast(unsigned short, h);
}

// A fragment (16-bit A 16x32): lane holds row mbase+(lane&15);
// elem e -> K = kc*32 + (lane>>4)*16 + e.
__device__ __forceinline__ void load_afrag(const float* __restrict__ A,
                                           int row, int kc, Frag& f) {
  const int lane = threadIdx.x & 31;
  const int kb = (lane >> 4) * 16;
  const float* p = A + (size_t)row * kH + kc * 32;
  const float4 a0 = *(const float4*)(p + kb);
  const float4 a1 = *(const float4*)(p + kb + 4);
  const float4 a2 = *(const float4*)(p + kb + 8);
  const float4 a3 = *(const float4*)(p + kb + 12);
  f.v[0]  = (__bf16)a0.x; f.v[1]  = (__bf16)a0.y;
  f.v[2]  = (__bf16)a0.z; f.v[3]  = (__bf16)a0.w;
  f.v[4]  = (__bf16)a1.x; f.v[5]  = (__bf16)a1.y;
  f.v[6]  = (__bf16)a1.z; f.v[7]  = (__bf16)a1.w;
  f.v[8]  = (__bf16)a2.x; f.v[9]  = (__bf16)a2.y;
  f.v[10] = (__bf16)a2.z; f.v[11] = (__bf16)a2.w;
  f.v[12] = (__bf16)a3.x; f.v[13] = (__bf16)a3.y;
  f.v[14] = (__bf16)a3.z; f.v[15] = (__bf16)a3.w;
}

// One wave accumulates a 16x128 strip: 4 k-chunks x 8 n-tiles of WMMA.
__device__ __forceinline__ void gemm_accum(const float* __restrict__ A, int mbase,
                                           const unsigned short* __restrict__ fr,
                                           v8f acc[8]) {
  const int lane = threadIdx.x & 31;
  const int row  = mbase + (lane & 15);
#pragma unroll
  for (int kc = 0; kc < 4; ++kc) {
    Frag af;
    load_afrag(A, row, kc, af);
#pragma unroll
    for (int nt = 0; nt < 8; ++nt) {
      Frag bf;
      const uint4* bq = (const uint4*)(fr + (((kc * 8 + nt) * 32 + lane) << 4));
      *(uint4*)&bf.u[0] = bq[0];
      *(uint4*)&bf.u[4] = bq[1];
      acc[nt] = __builtin_amdgcn_wmma_f32_16x16x32_bf16(
          false, af.v, false, bf.v, (short)0, acc[nt], false, false);
    }
  }
}

__device__ __forceinline__ void zero_acc(v8f acc[8]) {
  const v8f z = {0.f, 0.f, 0.f, 0.f, 0.f, 0.f, 0.f, 0.f};
#pragma unroll
  for (int nt = 0; nt < 8; ++nt) acc[nt] = z;
}

// C layout: VGPR r -> row mbase + r + 8*(lane>>4), column nt*16 + (lane&15).

__global__ void __launch_bounds__(256)
gemm_plain(const float* __restrict__ A, const unsigned short* __restrict__ fr,
           float* __restrict__ C, int M) {
  int strip = blockIdx.x * 8 + (threadIdx.x >> 5);
  if (strip * 16 >= M) return;
  int mbase = strip * 16;
  v8f acc[8]; zero_acc(acc);
  gemm_accum(A, mbase, fr, acc);
  const int lane = threadIdx.x & 31, h = lane >> 4, ln = lane & 15;
#pragma unroll
  for (int nt = 0; nt < 8; ++nt)
#pragma unroll
    for (int r = 0; r < 8; ++r)
      C[(size_t)(mbase + r + 8 * h) * kH + nt * 16 + ln] = acc[nt][r];
}

// C = LN(relu(A@W + bias + u (outer) v)) fused in one wave (full rows local).
__global__ void __launch_bounds__(256)
gemm_relu_ln_rank1(const float* __restrict__ A, const unsigned short* __restrict__ fr,
                   const float* __restrict__ bias, const float* __restrict__ uvec,
                   const float* __restrict__ vvec, const float* __restrict__ lng,
                   const float* __restrict__ lnb, float* __restrict__ C, int M) {
  int strip = blockIdx.x * 8 + (threadIdx.x >> 5);
  if (strip * 16 >= M) return;
  int mbase = strip * 16;
  v8f acc[8]; zero_acc(acc);
  gemm_accum(A, mbase, fr, acc);
  const int lane = threadIdx.x & 31, h = lane >> 4, ln = lane & 15;
  float u[8];
#pragma unroll
  for (int r = 0; r < 8; ++r) u[r] = uvec[mbase + r + 8 * h];
  float sum[8], sq[8];
#pragma unroll
  for (int r = 0; r < 8; ++r) { sum[r] = 0.f; sq[r] = 0.f; }
#pragma unroll
  for (int nt = 0; nt < 8; ++nt) {
    const int col = nt * 16 + ln;
    const float bn = bias[col], vn = vvec[col];
#pragma unroll
    for (int r = 0; r < 8; ++r) {
      float x = fmaxf(acc[nt][r] + bn + u[r] * vn, 0.f);
      sum[r] += x; sq[r] += x * x;
    }
  }
  // Reduce across the 16-lane half-group (rows r live in lanes 0-15, rows 8+r
  // in lanes 16-31; masks 1..8 stay within the half-group).
#pragma unroll
  for (int r = 0; r < 8; ++r)
#pragma unroll
    for (int m = 1; m < 16; m <<= 1) {
      sum[r] += __shfl_xor(sum[r], m, 32);
      sq[r]  += __shfl_xor(sq[r],  m, 32);
    }
  float mean[8], rstd[8];
#pragma unroll
  for (int r = 0; r < 8; ++r) {
    mean[r] = sum[r] * (1.f / kH);
    float var = sq[r] * (1.f / kH) - mean[r] * mean[r];
    rstd[r] = rsqrtf(var + kLnEps);
  }
#pragma unroll
  for (int nt = 0; nt < 8; ++nt) {
    const int col = nt * 16 + ln;
    const float bn = bias[col], vn = vvec[col];
    const float gg = lng[col], bb = lnb[col];
#pragma unroll
    for (int r = 0; r < 8; ++r) {
      float x = fmaxf(acc[nt][r] + bn + u[r] * vn, 0.f);
      C[(size_t)(mbase + r + 8 * h) * kH + col] = (x - mean[r]) * rstd[r] * gg + bb;
    }
  }
}

// out = sigmoid(X1@Wa + X2@Wb + bias) * X2 + (1-g) * X1 ; in-place safe
// (each strip's rows touched by exactly one wave).
__global__ void __launch_bounds__(256)
gemm_gate(const float* __restrict__ X1, const float* __restrict__ X2,
          const unsigned short* __restrict__ frA, const unsigned short* __restrict__ frB,
          const float* __restrict__ bias, float* __restrict__ out, int M) {
  int strip = blockIdx.x * 8 + (threadIdx.x >> 5);
  if (strip * 16 >= M) return;
  int mbase = strip * 16;
  v8f acc[8]; zero_acc(acc);
  gemm_accum(X1, mbase, frA, acc);
  gemm_accum(X2, mbase, frB, acc);
  const int lane = threadIdx.x & 31, h = lane >> 4, ln = lane & 15;
#pragma unroll
  for (int nt = 0; nt < 8; ++nt) {
    const int col = nt * 16 + ln;
    const float bn = bias[col];
#pragma unroll
    for (int r = 0; r < 8; ++r) {
      const size_t idx = (size_t)(mbase + r + 8 * h) * kH + col;
      float g = 1.f / (1.f + __expf(-(acc[nt][r] + bn)));
      float x1 = X1[idx], x2 = X2[idx];
      out[idx] = g * x2 + (1.f - g) * x1;
    }
  }
}

// One wave per edge: gather 128 floats of transformed source node, atomic-add
// into destination accumulator (all L2-resident).
__global__ void __launch_bounds__(256)
scatter_add(const float* __restrict__ msg, const int* __restrict__ src,
            const int* __restrict__ dst, float* __restrict__ acc, int E) {
  int e = blockIdx.x * 8 + (threadIdx.x >> 5);
  if (e >= E) return;
  const int lane = threadIdx.x & 31;
  const int sN = src[e], dN = dst[e];
  const float4 val = *(const float4*)(msg + (size_t)sN * kH + lane * 4);
  float* p = acc + (size_t)dN * kH + lane * 4;
  atomicAdd(p + 0, val.x);
  atomicAdd(p + 1, val.y);
  atomicAdd(p + 2, val.z);
  atomicAdd(p + 3, val.w);
}

__global__ void count_deg(const int* __restrict__ dst, float* __restrict__ deg, int E) {
  int e = blockIdx.x * blockDim.x + threadIdx.x;
  if (e < E) atomicAdd(&deg[dst[e]], 1.0f);
}

// agg = (acc + deg*bias) / (deg + eps)   (folds per-edge msg bias into mean)
__global__ void agg_norm(float* __restrict__ acc, const float* __restrict__ deg,
                         const float* __restrict__ bias, int M) {
  int i = blockIdx.x * blockDim.x + threadIdx.x;
  if (i >= M * kH) return;
  int row = i >> 7, col = i & 127;
  float d = deg[row];
  acc[i] = (acc[i] + d * bias[col]) / (d + kCntEps);
}

// Full-wave (32-lane) row LayerNorm; lane handles cols lane + 32*j.
__device__ __forceinline__ void rowln_store(const float v[4],
                                            const float* __restrict__ g,
                                            const float* __restrict__ b,
                                            float* __restrict__ out,
                                            size_t rowoff, int lane) {
  float s  = v[0] + v[1] + v[2] + v[3];
  float sq = v[0] * v[0] + v[1] * v[1] + v[2] * v[2] + v[3] * v[3];
#pragma unroll
  for (int m = 1; m < 32; m <<= 1) {
    s  += __shfl_xor(s,  m, 32);
    sq += __shfl_xor(sq, m, 32);
  }
  float mean = s * (1.f / kH);
  float var  = sq * (1.f / kH) - mean * mean;
  float rstd = rsqrtf(var + kLnEps);
#pragma unroll
  for (int j = 0; j < 4; ++j) {
    int c = lane + 32 * j;
    out[rowoff + c] = (v[j] - mean) * rstd * g[c] + b[c];
  }
}

__global__ void __launch_bounds__(256)
init_hv_kernel(const float* __restrict__ x, const float* __restrict__ w,
               const float* __restrict__ b, const float* __restrict__ lng,
               const float* __restrict__ lnb, float* __restrict__ out, int M) {
  int row = blockIdx.x * 8 + (threadIdx.x >> 5);
  if (row >= M) return;
  const int lane = threadIdx.x & 31;
  float xv[4];
#pragma unroll
  for (int d = 0; d < 4; ++d) xv[d] = x[row * 4 + d];
  float v[4];
#pragma unroll
  for (int j = 0; j < 4; ++j) {
    int c = lane + 32 * j;
    float p = b[c];
#pragma unroll
    for (int d = 0; d < 4; ++d) p += xv[d] * w[d * kH + c];
    v[j] = fmaxf(p, 0.f);
  }
  rowln_store(v, lng, lnb, out, (size_t)row * kH, lane);
}

__global__ void __launch_bounds__(256)
init_hc_kernel(const float* __restrict__ x, const float* __restrict__ w,
               const float* __restrict__ b, const float* __restrict__ lng,
               const float* __restrict__ lnb, float* __restrict__ out, int M) {
  int row = blockIdx.x * 8 + (threadIdx.x >> 5);
  if (row >= M) return;
  const int lane = threadIdx.x & 31;
  const float xv = x[row];
  float v[4];
#pragma unroll
  for (int j = 0; j < 4; ++j) {
    int c = lane + 32 * j;
    v[j] = fmaxf(xv * w[c] + b[c], 0.f);
  }
  rowln_store(v, lng, lnb, out, (size_t)row * kH, lane);
}

// var_new = LN(relu((acc + deg*bias)/(deg+eps))) in place, one wave per row.
__global__ void __launch_bounds__(256)
varnew_kernel(float* __restrict__ buf, const float* __restrict__ deg,
              const float* __restrict__ bias, const float* __restrict__ lng,
              const float* __restrict__ lnb, int M) {
  int row = blockIdx.x * 8 + (threadIdx.x >> 5);
  if (row >= M) return;
  const int lane = threadIdx.x & 31;
  const float d = deg[row];
  const float inv = 1.f / (d + kCntEps);
  float v[4];
#pragma unroll
  for (int j = 0; j < 4; ++j) {
    int c = lane + 32 * j;
    v[j] = fmaxf((buf[(size_t)row * kH + c] + d * bias[c]) * inv, 0.f);
  }
  rowln_store(v, lng, lnb, buf, (size_t)row * kH, lane);
}

extern "C" void kernel_launch(void* const* d_in, const int* in_sizes, int n_in,
                              void* d_out, int out_size, void* d_ws, size_t ws_size,
                              hipStream_t stream) {
  (void)in_sizes; (void)n_in; (void)out_size; (void)ws_size;
  const float* x_var      = (const float*)d_in[0];
  const float* x_con      = (const float*)d_in[1];
  const int*   e_vc_src   = (const int*)d_in[2];
  const int*   e_vc_dst   = (const int*)d_in[3];
  const int*   e_cv_src   = (const int*)d_in[4];
  const int*   e_cv_dst   = (const int*)d_in[5];
  const float* var_proj_w = (const float*)d_in[6];
  const float* var_proj_b = (const float*)d_in[7];
  const float* con_proj_w = (const float*)d_in[8];
  const float* con_proj_b = (const float*)d_in[9];
  const float* v2c_msg_w  = (const float*)d_in[10];
  const float* v2c_msg_b  = (const float*)d_in[11];
  const float* v2c_upd_w  = (const float*)d_in[12];
  const float* v2c_upd_b  = (const float*)d_in[13];
  const float* v2c_ln_g   = (const float*)d_in[14];
  const float* v2c_ln_b   = (const float*)d_in[15];
  const float* c2v_msg_w  = (const float*)d_in[16];
  const float* c2v_msg_b  = (const float*)d_in[17];
  const float* c2v_ln_g   = (const float*)d_in[18];
  const float* c2v_ln_b   = (const float*)d_in[19];
  const float* var_gate_w = (const float*)d_in[20];
  const float* var_gate_b = (const float*)d_in[21];
  const float* con_gate_w = (const float*)d_in[22];
  const float* con_gate_b = (const float*)d_in[23];
  const float* var_ln_g   = (const float*)d_in[24];
  const float* var_ln_b   = (const float*)d_in[25];
  const float* con_ln_g   = (const float*)d_in[26];
  const float* con_ln_b   = (const float*)d_in[27];

  float* hv    = (float*)d_out;                 // hv lives in d_out throughout
  float* ws    = (float*)d_ws;
  float* bufV  = ws;                            // NV*H : mv / accV / var_new
  float* hc    = bufV + (size_t)kNV * kH;       // NC*H
  float* bufC  = hc   + (size_t)kNC * kH;       // NC*H : accC / agg / con_new
  float* bufC2 = bufC + (size_t)kNC * kH;       // NC*H : mc
  float* degv  = bufC2 + (size_t)kNC * kH;      // NV
  float* degc  = degv + kNV;                    // NC
  unsigned short* fr = (unsigned short*)(degc + kNC);
  unsigned short* fr_v2c = fr + 0 * kFragElems;
  unsigned short* fr_upd = fr + 1 * kFragElems;
  unsigned short* fr_c2v = fr + 2 * kFragElems;
  unsigned short* fr_cga = fr + 3 * kFragElems;
  unsigned short* fr_cgb = fr + 4 * kFragElems;
  unsigned short* fr_vga = fr + 5 * kFragElems;
  unsigned short* fr_vgb = fr + 6 * kFragElems;

  // Pack weights into bf16 WMMA B-fragments (tiny, once per launch).
  prep_wfrag<<<64, 256, 0, stream>>>(v2c_msg_w,             fr_v2c);
  prep_wfrag<<<64, 256, 0, stream>>>(v2c_upd_w,             fr_upd);  // rows 0..127
  prep_wfrag<<<64, 256, 0, stream>>>(c2v_msg_w,             fr_c2v);
  prep_wfrag<<<64, 256, 0, stream>>>(con_gate_w,            fr_cga);  // rows 0..127
  prep_wfrag<<<64, 256, 0, stream>>>(con_gate_w + 128 * kH, fr_cgb);  // rows 128..255
  prep_wfrag<<<64, 256, 0, stream>>>(var_gate_w,            fr_vga);
  prep_wfrag<<<64, 256, 0, stream>>>(var_gate_w + 128 * kH, fr_vgb);

  hipMemsetAsync(degv, 0, (size_t)kNV * sizeof(float), stream);
  hipMemsetAsync(degc, 0, (size_t)kNC * sizeof(float), stream);
  count_deg<<<(kNE + 255) / 256, 256, 0, stream>>>(e_vc_dst, degc, kNE);
  count_deg<<<(kNE + 255) / 256, 256, 0, stream>>>(e_cv_dst, degv, kNE);

  init_hv_kernel<<<(kNV + 7) / 8, 256, 0, stream>>>(x_var, var_proj_w, var_proj_b,
                                                    var_ln_g, var_ln_b, hv, kNV);
  init_hc_kernel<<<(kNC + 7) / 8, 256, 0, stream>>>(x_con, con_proj_w, con_proj_b,
                                                    con_ln_g, con_ln_b, hc, kNC);

  const int gV = (kNV / 16 + 7) / 8;   // 782 blocks, 8 waves/block
  const int gC = (kNC / 16 + 7) / 8;   // 391 blocks
  const int gE = (kNE + 7) / 8;

  for (int round = 0; round < 2; ++round) {
    // v -> c : mv = hv @ v2c_msg_w ; segment mean ; update ; gate
    gemm_plain<<<gV, 256, 0, stream>>>(hv, fr_v2c, bufV, kNV);
    hipMemsetAsync(bufC, 0, (size_t)kNC * kH * sizeof(float), stream);
    scatter_add<<<gE, 256, 0, stream>>>(bufV, e_vc_src, e_vc_dst, bufC, kNE);
    agg_norm<<<(kNC * kH + 255) / 256, 256, 0, stream>>>(bufC, degc, v2c_msg_b, kNC);
    gemm_relu_ln_rank1<<<gC, 256, 0, stream>>>(bufC, fr_upd, v2c_upd_b,
                                               x_con /* clue */, v2c_upd_w + 128 * kH,
                                               v2c_ln_g, v2c_ln_b, bufC, kNC);
    gemm_gate<<<gC, 256, 0, stream>>>(hc, bufC, fr_cga, fr_cgb, con_gate_b, hc, kNC);

    // c -> v : mc = hc @ c2v_msg_w ; segment mean ; LN ; gate
    gemm_plain<<<gC, 256, 0, stream>>>(hc, fr_c2v, bufC2, kNC);
    hipMemsetAsync(bufV, 0, (size_t)kNV * kH * sizeof(float), stream);
    scatter_add<<<gE, 256, 0, stream>>>(bufC2, e_cv_src, e_cv_dst, bufV, kNE);
    varnew_kernel<<<(kNV + 7) / 8, 256, 0, stream>>>(bufV, degv, c2v_msg_b,
                                                     c2v_ln_g, c2v_ln_b, kNV);
    gemm_gate<<<gV, 256, 0, stream>>>(hv, bufV, fr_vga, fr_vgb, var_gate_b, hv, kNV);
  }
}
